// Performer_72430328480167
// MI455X (gfx1250) — compile-verified, matrix operand order checked by reading, not run on hardware
//
#include <hip/hip_runtime.h>
#include <hip/hip_bf16.h>
#include <math.h>

#define BATCH   8
#define NSEQ    4096
#define DIMX    3
#define NHEADS  3
#define DHEAD   128
#define INNERD  384
#define MFEAT   620
#define MP      640          // M padded to 40 tiles of 16
#define NMT     (MP/16)      // 40
#define FFD     12
#define DEPTH   6
#define BHX     (BATCH*NHEADS)

#define DNORM   0.29730177875068026f     // 128^-0.25
#define RATIOF  0.040160966445124896f    // 620^-0.5
#define EPSV    1e-4f
#define NEGBIG  -3.0e38f

typedef __bf16 bf16_t;
typedef bf16_t v16bf __attribute__((ext_vector_type(16)));
typedef bf16_t v8bf  __attribute__((ext_vector_type(8)));
typedef float  v8f   __attribute__((ext_vector_type(8)));

static __device__ __forceinline__ v8f wmma_bf16(v16bf a, v16bf b, v8f c) {
    return __builtin_amdgcn_wmma_f32_16x16x32_bf16(false, a, false, b, (short)0, c, false, false);
}

// Fragment loader for A (row r = lane&15, K contiguous) and B-from-transposed
// storage (row c = lane&15, K contiguous). Two 16B loads per lane.
static __device__ __forceinline__ v16bf load_frag(const bf16_t* p, int ld) {
    const int lane = threadIdx.x & 31;
    const int r  = lane & 15;
    const int k0 = (lane >> 4) << 3;
    const bf16_t* q = p + (size_t)r * ld + k0;
    v8bf lo = *(const v8bf*)(q);
    v8bf hi = *(const v8bf*)(q + 16);
    return __builtin_shufflevector(lo, hi, 0, 1, 2, 3, 4, 5, 6, 7,
                                           8, 9, 10, 11, 12, 13, 14, 15);
}

// monotone float<->int mapping for atomicMax on signed int
static __device__ __forceinline__ int   f2ord(float f) { int i = __float_as_int(f); return i >= 0 ? i : (i ^ 0x7fffffff); }
static __device__ __forceinline__ float ord2f(int i)   { return __int_as_float(i >= 0 ? i : (i ^ 0x7fffffff)); }

// ---------------------------------------------------------------- utility
__global__ void copy_x_kernel(const float* __restrict__ xin, float* __restrict__ xbuf, int n) {
    int t = blockIdx.x * blockDim.x + threadIdx.x;
    if (t < n) xbuf[t] = xin[t];
}

// proj[DEPTH][M][DH] (f32) -> projB[DEPTH][MP][DH] (bf16, rows m>=620 zero)
__global__ void prep_proj_kernel(const float* __restrict__ proj, bf16_t* __restrict__ projB) {
    int t = blockIdx.x * blockDim.x + threadIdx.x;
    const int total = DEPTH * MP * DHEAD;
    if (t >= total) return;
    int d = t % DHEAD; int r = t / DHEAD;
    int m = r % MP;    int l = r / MP;
    float v = (m < MFEAT) ? proj[((size_t)l * MFEAT + m) * DHEAD + d] : 0.0f;
    projB[((size_t)l * MP + m) * DHEAD + d] = (bf16_t)v;
}

__global__ void init_kmax_kernel(int* kmax) {
    int t = blockIdx.x * blockDim.x + threadIdx.x;
    if (t < BHX) kmax[t] = (int)0x80000000;
}

// ------------------------------------------------- LN1 + QKV (DIM=3 scalar)
__global__ void qkv_kernel(const float* __restrict__ xbuf,
                           const float* __restrict__ Wq, const float* __restrict__ Wk,
                           const float* __restrict__ Wv,
                           const float* __restrict__ g, const float* __restrict__ bb,
                           bf16_t* __restrict__ qo, bf16_t* __restrict__ ko, bf16_t* __restrict__ vT,
                           float* __restrict__ diag_q, float* __restrict__ diag_k) {
    __shared__ float sq[INNERD];
    __shared__ float sk[INNERD];
    const int token = blockIdx.x;             // b*NSEQ+n
    const int t = threadIdx.x;                // 0..383
    const int b = token >> 12, n = token & (NSEQ - 1);
    const float x0 = xbuf[(size_t)token * 3 + 0];
    const float x1 = xbuf[(size_t)token * 3 + 1];
    const float x2 = xbuf[(size_t)token * 3 + 2];
    const float mu = (x0 + x1 + x2) * (1.0f / 3.0f);
    const float v0 = x0 - mu, v1 = x1 - mu, v2 = x2 - mu;
    const float var = (v0 * v0 + v1 * v1 + v2 * v2) * (1.0f / 3.0f);
    const float rs = rsqrtf(var + 1e-5f);
    const float h0 = v0 * rs * g[0] + bb[0];
    const float h1 = v1 * rs * g[1] + bb[1];
    const float h2 = v2 * rs * g[2] + bb[2];
    const float qv = h0 * Wq[t] + h1 * Wq[INNERD + t] + h2 * Wq[2 * INNERD + t];
    const float kv = h0 * Wk[t] + h1 * Wk[INNERD + t] + h2 * Wk[2 * INNERD + t];
    const float vv = h0 * Wv[t] + h1 * Wv[INNERD + t] + h2 * Wv[2 * INNERD + t];
    const int head = t >> 7, d = t & 127;
    const float qs = qv * DNORM, ks = kv * DNORM;
    const int bh = b * NHEADS + head;
    const size_t idx = ((size_t)bh * NSEQ + n) * DHEAD + d;
    qo[idx] = (bf16_t)qs; ko[idx] = (bf16_t)ks;
    vT[((size_t)bh * DHEAD + d) * NSEQ + n] = (bf16_t)vv;   // transposed store
    sq[t] = qs * qs; sk[t] = ks * ks;
    __syncthreads();
    if (d == 0) {
        float s1 = 0.f, s2 = 0.f;
        for (int i = 0; i < DHEAD; ++i) { s1 += sq[head * DHEAD + i]; s2 += sk[head * DHEAD + i]; }
        diag_q[(size_t)bh * NSEQ + n] = 0.5f * s1;
        diag_k[(size_t)bh * NSEQ + n] = 0.5f * s2;
    }
}

// ------------- pass 1 over k: global max of dd_k (2 n-tiles per wave,
// double-buffered B, branch-free masking)
__global__ void __launch_bounds__(32) kmax_kernel(const bf16_t* __restrict__ kq,
                                                  const bf16_t* __restrict__ projB_l,
                                                  int* __restrict__ kmax) {
    const int nt = blockIdx.x, bh = blockIdx.y, lane = threadIdx.x;
    const bf16_t* kb = kq + ((size_t)bh * NSEQ + nt * 32) * DHEAD;
    const int col = lane & 15;
    v16bf a0[4], a1[4];
#pragma unroll
    for (int kk = 0; kk < 4; ++kk) {
        a0[kk] = load_frag(kb + kk * 32, DHEAD);
        a1[kk] = load_frag(kb + (size_t)16 * DHEAD + kk * 32, DHEAD);
    }
    float mx = NEGBIG;
    v16bf bufA[4], bufB[4];
#pragma unroll
    for (int kk = 0; kk < 4; ++kk) bufA[kk] = load_frag(projB_l + kk * 32, DHEAD);

    auto step = [&](v16bf (&bcur)[4], v16bf (&bnxt)[4], int mt) {
        if (mt + 1 < NMT) {
#pragma unroll
            for (int kk = 0; kk < 4; ++kk)
                bnxt[kk] = load_frag(projB_l + (size_t)((mt + 1) * 16) * DHEAD + kk * 32, DHEAD);
        }
        v8f c0 = {0.f,0.f,0.f,0.f,0.f,0.f,0.f,0.f}, c1 = c0;
#pragma unroll
        for (int kk = 0; kk < 4; ++kk) {
            c0 = wmma_bf16(a0[kk], bcur[kk], c0);
            c1 = wmma_bf16(a1[kk], bcur[kk], c1);
        }
        float mxt = NEGBIG;
#pragma unroll
        for (int j = 0; j < 8; ++j) { mxt = fmaxf(mxt, c0[j]); mxt = fmaxf(mxt, c1[j]); }
        mx = fmaxf(mx, (mt * 16 + col < MFEAT) ? mxt : NEGBIG);  // one select, no exec branch
    };
    for (int mt = 0; mt < NMT; mt += 2) { step(bufA, bufB, mt); step(bufB, bufA, mt + 1); }

    for (int off = 16; off; off >>= 1) mx = fmaxf(mx, __shfl_xor(mx, off, 32));
    if (lane == 0) atomicMax(&kmax[bh], f2ord(mx));
}

// ------ pass 2 over k: kp -> kpT[bh][m][n] (2 n-tiles per wave, dbl-buffer)
__global__ void __launch_bounds__(32) kp_kernel(const bf16_t* __restrict__ kq,
                                                const bf16_t* __restrict__ projB_l,
                                                const float* __restrict__ diag_k,
                                                const int* __restrict__ kmax,
                                                bf16_t* __restrict__ kpT) {
    const int nt = blockIdx.x, bh = blockIdx.y, lane = threadIdx.x;
    const bf16_t* kb = kq + ((size_t)bh * NSEQ + nt * 32) * DHEAD;
    const int rowo = (lane >> 4) << 3, col = lane & 15;
    v16bf a0[4], a1[4];
#pragma unroll
    for (int kk = 0; kk < 4; ++kk) {
        a0[kk] = load_frag(kb + kk * 32, DHEAD);
        a1[kk] = load_frag(kb + (size_t)16 * DHEAD + kk * 32, DHEAD);
    }
    const float kmx = ord2f(kmax[bh]);
    float dg0[8], dg1[8];
#pragma unroll
    for (int j = 0; j < 8; ++j) {
        dg0[j] = diag_k[(size_t)bh * NSEQ + nt * 32 + rowo + j];
        dg1[j] = diag_k[(size_t)bh * NSEQ + nt * 32 + 16 + rowo + j];
    }
    v16bf bufA[4], bufB[4];
#pragma unroll
    for (int kk = 0; kk < 4; ++kk) bufA[kk] = load_frag(projB_l + kk * 32, DHEAD);

    auto step = [&](v16bf (&bcur)[4], v16bf (&bnxt)[4], int mt) {
        if (mt + 1 < NMT) {
#pragma unroll
            for (int kk = 0; kk < 4; ++kk)
                bnxt[kk] = load_frag(projB_l + (size_t)((mt + 1) * 16) * DHEAD + kk * 32, DHEAD);
        }
        v8f c0 = {0.f,0.f,0.f,0.f,0.f,0.f,0.f,0.f}, c1 = c0;
#pragma unroll
        for (int kk = 0; kk < 4; ++kk) {
            c0 = wmma_bf16(a0[kk], bcur[kk], c0);
            c1 = wmma_bf16(a1[kk], bcur[kk], c1);
        }
        const int m = mt * 16 + col;
        const bool valid = m < MFEAT;
        v8bf p0, p1;
#pragma unroll
        for (int j = 0; j < 8; ++j) {
            float v0 = valid ? RATIOF * (expf(c0[j] - dg0[j] - kmx) + EPSV) : 0.0f;
            float v1 = valid ? RATIOF * (expf(c1[j] - dg1[j] - kmx) + EPSV) : 0.0f;
            p0[j] = (bf16_t)v0; p1[j] = (bf16_t)v1;
        }
        bf16_t* dst = kpT + ((size_t)bh * MP + m) * NSEQ + nt * 32 + rowo;
        *(v8bf*)(dst) = p0;
        *(v8bf*)(dst + 16) = p1;
    };
    for (int mt = 0; mt < NMT; mt += 2) { step(bufA, bufB, mt); step(bufB, bufA, mt + 1); }
}

// k_sum[bh][m] = sum_n kpT[bh][m][n]  (wave per row, deterministic)
__global__ void __launch_bounds__(32) ksum_kernel(const bf16_t* __restrict__ kpT, float* __restrict__ ksum) {
    const int row = blockIdx.x;                  // bh*MP + m
    const int lane = threadIdx.x;
    const bf16_t* p = kpT + (size_t)row * NSEQ;
    float s = 0.f;
    for (int n = lane * 8; n < NSEQ; n += 32 * 8) {
        v8bf v = *(const v8bf*)(p + n);
#pragma unroll
        for (int j = 0; j < 8; ++j) s += (float)v[j];
    }
    for (int off = 16; off; off >>= 1) s += __shfl_xor(s, off, 32);
    if (lane == 0) ksum[row] = s;
}

// ---- ctxT[d][m] = sum_n kpT[m][n] * vT[d][n]  (2x2 blocking, dbl-buffer)
__global__ void __launch_bounds__(32) ctx_kernel(const bf16_t* __restrict__ kpT,
                                                 const bf16_t* __restrict__ vT,
                                                 bf16_t* __restrict__ ctxT) {
    const int tile = blockIdx.x, bh = blockIdx.y;     // tile = mt*(DHEAD/32)+dt
    const int mt = tile >> 2, dt = tile & 3;          // m0=mt*32, d0=dt*32
    const bf16_t* ap = kpT + ((size_t)bh * MP + mt * 32) * NSEQ;
    const bf16_t* bp = vT + ((size_t)bh * DHEAD + dt * 32) * NSEQ;
    v8f c00 = {0.f,0.f,0.f,0.f,0.f,0.f,0.f,0.f}, c01 = c00, c10 = c00, c11 = c00;
    v16bf bufA[4], bufB[4];
    bufA[0] = load_frag(ap, NSEQ);
    bufA[1] = load_frag(ap + (size_t)16 * NSEQ, NSEQ);
    bufA[2] = load_frag(bp, NSEQ);
    bufA[3] = load_frag(bp + (size_t)16 * NSEQ, NSEQ);

    auto step = [&](v16bf (&cur)[4], v16bf (&nxt)[4], int n0) {
        if (n0 + 32 < NSEQ) {
            nxt[0] = load_frag(ap + n0 + 32, NSEQ);
            nxt[1] = load_frag(ap + (size_t)16 * NSEQ + n0 + 32, NSEQ);
            nxt[2] = load_frag(bp + n0 + 32, NSEQ);
            nxt[3] = load_frag(bp + (size_t)16 * NSEQ + n0 + 32, NSEQ);
        }
        if (n0 + 256 < NSEQ) {
            __builtin_prefetch(ap + n0 + 256, 0, 1);
            __builtin_prefetch(bp + n0 + 256, 0, 1);
        }
        c00 = wmma_bf16(cur[0], cur[2], c00);
        c01 = wmma_bf16(cur[0], cur[3], c01);
        c10 = wmma_bf16(cur[1], cur[2], c10);
        c11 = wmma_bf16(cur[1], cur[3], c11);
    };
    for (int n0 = 0; n0 < NSEQ; n0 += 64) { step(bufA, bufB, n0); step(bufB, bufA, n0 + 32); }

    const int lane = threadIdx.x, rowo = (lane >> 4) << 3, col = lane & 15;
    bf16_t* ob = ctxT + (size_t)bh * DHEAD * MP;
    const v8f* cc[4] = { &c00, &c01, &c10, &c11 };
#pragma unroll
    for (int mi = 0; mi < 2; ++mi)
#pragma unroll
        for (int dj = 0; dj < 2; ++dj) {
            const v8f& c = *cc[mi * 2 + dj];
            v8bf pk;
#pragma unroll
            for (int j = 0; j < 8; ++j) pk[j] = (bf16_t)c[j];
            *(v8bf*)(ob + (size_t)(dt * 32 + dj * 16 + col) * MP + mt * 32 + mi * 16 + rowo) = pk;
        }
}

// ---- fused q side: dd_q -> rowmax -> qp(LDS) -> denom -> qp@ctx -> out_h
__global__ void __launch_bounds__(32) qout_kernel(const bf16_t* __restrict__ q,
                                                  const bf16_t* __restrict__ projB_l,
                                                  const float* __restrict__ diag_q,
                                                  const float* __restrict__ ksum,
                                                  const bf16_t* __restrict__ ctxT,
                                                  float* __restrict__ outh) {
    __shared__ __align__(16) float  dd_ls[MP * 16];     // [m][r] transposed
    __shared__ __align__(16) bf16_t qp_ls[16 * MP];     // [r][m]
    __shared__ float row_mx[16];
    __shared__ float row_dg[16];
    __shared__ float row_dn[16];
    const int nt = blockIdx.x, bh = blockIdx.y, lane = threadIdx.x;
    const int b = bh / NHEADS, h = bh % NHEADS;
    const bf16_t* qb = q + ((size_t)bh * NSEQ + nt * 16) * DHEAD;
    const int rowo = (lane >> 4) << 3, col = lane & 15;
    v16bf afr[4];
#pragma unroll
    for (int kk = 0; kk < 4; ++kk) afr[kk] = load_frag(qb + kk * 32, DHEAD);
    // phase 1: dd_q tiles into LDS (double-buffered B)
    v16bf bufA[4], bufB[4];
#pragma unroll
    for (int kk = 0; kk < 4; ++kk) bufA[kk] = load_frag(projB_l + kk * 32, DHEAD);
    auto p1step = [&](v16bf (&bcur)[4], v16bf (&bnxt)[4], int mt) {
        if (mt + 1 < NMT) {
#pragma unroll
            for (int kk = 0; kk < 4; ++kk)
                bnxt[kk] = load_frag(projB_l + (size_t)((mt + 1) * 16) * DHEAD + kk * 32, DHEAD);
        }
        v8f c = {0.f,0.f,0.f,0.f,0.f,0.f,0.f,0.f};
#pragma unroll
        for (int kk = 0; kk < 4; ++kk) c = wmma_bf16(afr[kk], bcur[kk], c);
        float* dst = &dd_ls[(mt * 16 + col) * 16 + rowo];
#pragma unroll
        for (int j = 0; j < 8; ++j) dst[j] = c[j];
    };
    for (int mt = 0; mt < NMT; mt += 2) { p1step(bufA, bufB, mt); p1step(bufB, bufA, mt + 1); }

    if (lane < 16) row_dg[lane] = diag_q[(size_t)bh * NSEQ + nt * 16 + lane];
    __syncthreads();
    // phase 2: per-row max over real features
    if (lane < 16) {
        float mx = NEGBIG;
        for (int m = 0; m < MFEAT; ++m) mx = fmaxf(mx, dd_ls[m * 16 + lane]);
        row_mx[lane] = mx;
    }
    __syncthreads();
    // qp into LDS (bf16), zero-padded features
    for (int r = 0; r < 16; ++r) {
        const float sub = row_dg[r] + row_mx[r];
        for (int m = lane; m < MP; m += 32) {
            float val = (m < MFEAT) ? RATIOF * (expf(dd_ls[m * 16 + r] - sub) + EPSV) : 0.0f;
            qp_ls[r * MP + m] = (bf16_t)val;
        }
    }
    __syncthreads();
    // denominator 1/(qp . k_sum)
    if (lane < 16) {
        const float* ks = ksum + (size_t)bh * MP;
        float s = 0.f;
        for (int m = 0; m < MFEAT; ++m) s += (float)qp_ls[lane * MP + m] * ks[m];
        row_dn[lane] = 1.0f / s;
    }
    __syncthreads();
    // phase 3: out = (qp @ ctx) * d_inv  (1x2 dt blocking, dbl-buffered B)
    const bf16_t* cb = ctxT + (size_t)bh * DHEAD * MP;
    float* ob = outh + ((size_t)b * NSEQ + nt * 16) * INNERD + h * DHEAD;
    for (int dtb = 0; dtb < 4; ++dtb) {
        v8f c0 = {0.f,0.f,0.f,0.f,0.f,0.f,0.f,0.f}, c1 = c0;
        v16bf pbA[2], pbB[2];
        pbA[0] = load_frag(cb + (size_t)(dtb * 32) * MP, MP);
        pbA[1] = load_frag(cb + (size_t)(dtb * 32 + 16) * MP, MP);
        auto p3step = [&](v16bf (&bcur)[2], v16bf (&bnxt)[2], int kk) {
            if (kk + 32 < MP) {
                bnxt[0] = load_frag(cb + (size_t)(dtb * 32) * MP + kk + 32, MP);
                bnxt[1] = load_frag(cb + (size_t)(dtb * 32 + 16) * MP + kk + 32, MP);
            }
            v16bf a = load_frag(qp_ls + kk, MP);
            c0 = wmma_bf16(a, bcur[0], c0);
            c1 = wmma_bf16(a, bcur[1], c1);
        };
        for (int kk = 0; kk < MP; kk += 64) { p3step(pbA, pbB, kk); p3step(pbB, pbA, kk + 32); }
#pragma unroll
        for (int j = 0; j < 8; ++j) {
            const float dn = row_dn[rowo + j];
            ob[(size_t)(rowo + j) * INNERD + dtb * 32 + col] = c0[j] * dn;
            ob[(size_t)(rowo + j) * INNERD + dtb * 32 + 16 + col] = c1[j] * dn;
        }
    }
}

// ------------- Wo + residual + LN2 + FFN(GELU exact) + residual (scalar)
__global__ void __launch_bounds__(32) post_kernel(float* __restrict__ xbuf,
                                                  const float* __restrict__ outh,
                                                  const float* __restrict__ Wo, const float* __restrict__ bo,
                                                  const float* __restrict__ g2, const float* __restrict__ b2g,
                                                  const float* __restrict__ W1, const float* __restrict__ b1,
                                                  const float* __restrict__ W2, const float* __restrict__ b2) {
    const int token = blockIdx.x, lane = threadIdx.x;
    const float* oh = outh + (size_t)token * INNERD;
    float a0 = 0.f, a1 = 0.f, a2 = 0.f;
    for (int t = lane; t < INNERD; t += 32) {
        const float ov = oh[t];
        a0 += ov * Wo[t * 3 + 0]; a1 += ov * Wo[t * 3 + 1]; a2 += ov * Wo[t * 3 + 2];
    }
    for (int off = 16; off; off >>= 1) {
        a0 += __shfl_xor(a0, off, 32); a1 += __shfl_xor(a1, off, 32); a2 += __shfl_xor(a2, off, 32);
    }
    if (lane == 0) {
        float xv[3];
        xv[0] = xbuf[(size_t)token * 3 + 0] + a0 + bo[0];
        xv[1] = xbuf[(size_t)token * 3 + 1] + a1 + bo[1];
        xv[2] = xbuf[(size_t)token * 3 + 2] + a2 + bo[2];
        const float mu = (xv[0] + xv[1] + xv[2]) * (1.0f / 3.0f);
        const float d0 = xv[0] - mu, d1 = xv[1] - mu, d2 = xv[2] - mu;
        const float rs = rsqrtf((d0 * d0 + d1 * d1 + d2 * d2) * (1.0f / 3.0f) + 1e-5f);
        float hh[3] = { d0 * rs * g2[0] + b2g[0], d1 * rs * g2[1] + b2g[1], d2 * rs * g2[2] + b2g[2] };
        float f[FFD];
        for (int j = 0; j < FFD; ++j) {
            float s = b1[j];
            for (int c = 0; c < 3; ++c) s += hh[c] * W1[c * FFD + j];
            f[j] = 0.5f * s * (1.0f + erff(s * 0.70710678118654752f));
        }
        for (int c = 0; c < 3; ++c) {
            float s = b2[c];
            for (int j = 0; j < FFD; ++j) s += f[j] * W2[j * 3 + c];
            xbuf[(size_t)token * 3 + c] = xv[c] + s;
        }
    }
}

__global__ void dec_kernel(const float* __restrict__ xbuf, const float* __restrict__ dw,
                           const float* __restrict__ db, float* __restrict__ out, int n) {
    int t = blockIdx.x * blockDim.x + threadIdx.x;
    if (t >= n) return;
    float s = db[0];
    for (int c = 0; c < 3; ++c) s += xbuf[(size_t)t * 3 + c] * dw[c];
    out[t] = 1.0f / (1.0f + expf(-s));
}

// ------------------------------------------------------------------ host
extern "C" void kernel_launch(void* const* d_in, const int* in_sizes, int n_in,
                              void* d_out, int out_size, void* d_ws, size_t ws_size,
                              hipStream_t stream) {
    const float* x_in = (const float*)d_in[0];
    const float* Wq   = (const float*)d_in[1];
    const float* Wk   = (const float*)d_in[2];
    const float* Wv   = (const float*)d_in[3];
    const float* Wo   = (const float*)d_in[4];
    const float* bo   = (const float*)d_in[5];
    const float* ln1g = (const float*)d_in[6];
    const float* ln1b = (const float*)d_in[7];
    const float* W1   = (const float*)d_in[8];
    const float* b1   = (const float*)d_in[9];
    const float* W2   = (const float*)d_in[10];
    const float* b2   = (const float*)d_in[11];
    const float* ln2g = (const float*)d_in[12];
    const float* ln2b = (const float*)d_in[13];
    const float* proj = (const float*)d_in[14];
    const float* dw   = (const float*)d_in[15];
    const float* db   = (const float*)d_in[16];

    char* w = (char*)d_ws;
    size_t off = 0;
    auto carve = [&](size_t bytes) { char* p = w + off; off = (off + bytes + 255) & ~(size_t)255; return p; };
    float*  xbuf   = (float*)carve((size_t)BATCH * NSEQ * DIMX * 4);
    bf16_t* projB  = (bf16_t*)carve((size_t)DEPTH * MP * DHEAD * 2);
    bf16_t* qb     = (bf16_t*)carve((size_t)BHX * NSEQ * DHEAD * 2);
    bf16_t* kb     = (bf16_t*)carve((size_t)BHX * NSEQ * DHEAD * 2);
    bf16_t* vT     = (bf16_t*)carve((size_t)BHX * DHEAD * NSEQ * 2);
    float*  diag_q = (float*)carve((size_t)BHX * NSEQ * 4);
    float*  diag_k = (float*)carve((size_t)BHX * NSEQ * 4);
    int*    kmaxp  = (int*)carve((size_t)BHX * 4);
    float*  ksum   = (float*)carve((size_t)BHX * MP * 4);
    bf16_t* kpT    = (bf16_t*)carve((size_t)BHX * MP * NSEQ * 2);
    bf16_t* ctxT   = (bf16_t*)carve((size_t)BHX * DHEAD * MP * 2);
    float*  outh   = (float*)carve((size_t)BATCH * NSEQ * INNERD * 4);
    (void)ws_size; (void)n_in; (void)in_sizes; (void)out_size;

    const int nx = BATCH * NSEQ * DIMX;
    copy_x_kernel<<<(nx + 255) / 256, 256, 0, stream>>>(x_in, xbuf, nx);
    const int np = DEPTH * MP * DHEAD;
    prep_proj_kernel<<<(np + 255) / 256, 256, 0, stream>>>(proj, projB);

    for (int l = 0; l < DEPTH; ++l) {
        const bf16_t* pB = projB + (size_t)l * MP * DHEAD;
        init_kmax_kernel<<<1, 32, 0, stream>>>(kmaxp);
        qkv_kernel<<<BATCH * NSEQ, INNERD, 0, stream>>>(
            xbuf, Wq + (size_t)l * DIMX * INNERD, Wk + (size_t)l * DIMX * INNERD,
            Wv + (size_t)l * DIMX * INNERD, ln1g + l * DIMX, ln1b + l * DIMX,
            qb, kb, vT, diag_q, diag_k);
        dim3 gtile2(NSEQ / 32, BHX);
        kmax_kernel<<<gtile2, 32, 0, stream>>>(kb, pB, kmaxp);
        kp_kernel<<<gtile2, 32, 0, stream>>>(kb, pB, diag_k, kmaxp, kpT);
        ksum_kernel<<<BHX * MP, 32, 0, stream>>>(kpT, ksum);
        dim3 gctx((MP / 32) * (DHEAD / 32), BHX);
        ctx_kernel<<<gctx, 32, 0, stream>>>(kpT, vT, ctxT);
        dim3 gtile(NSEQ / 16, BHX);
        qout_kernel<<<gtile, 32, 0, stream>>>(qb, pB, diag_q, ksum, ctxT, outh);
        post_kernel<<<BATCH * NSEQ, 32, 0, stream>>>(
            xbuf, outh, Wo + (size_t)l * INNERD * DIMX, bo + l * DIMX,
            ln2g + l * DIMX, ln2b + l * DIMX, W1 + (size_t)l * DIMX * FFD, b1 + l * FFD,
            W2 + (size_t)l * FFD * DIMX, b2 + l * DIMX);
    }
    const int nt = BATCH * NSEQ;
    dec_kernel<<<(nt + 255) / 256, 256, 0, stream>>>(xbuf, dw, db, (float*)d_out, nt);
}
